// SwinTransformerBlock_42674795053152
// MI455X (gfx1250) — compile-verified, matrix operand order checked by reading, not run on hardware
//
#include <hip/hip_runtime.h>
#include <math.h>

typedef __bf16 bf16_t;
typedef __attribute__((ext_vector_type(16))) __bf16 v16bf;
typedef __attribute__((ext_vector_type(8)))  float  v8f;

#define TOK     100352          // 2048 windows * 49 tokens
#define NWIN    2048
#define DIMC    192
#define HEADS   6
#define HD      32              // head dim
#define NPAD    64              // padded tokens per window
#define SCALE_Q 0.17677669529663689f

// ---------- WMMA helpers ----------

__device__ __forceinline__ v8f wmma_bf16(v16bf a, v16bf b, v8f c) {
  return __builtin_amdgcn_wmma_f32_16x16x32_bf16(false, a, false, b, (short)0, c, false, false);
}

// Load a 16x32 bf16 tile in WMMA A/B per-lane layout from a row-major matrix
// (row stride ld elements). lane<16: row=lane, K {0..7},{16..23};
// lane>=16: row=lane-16, K {8..15},{24..31}. Two 16B loads per lane.
__device__ __forceinline__ v16bf load_tile16x32(const bf16_t* __restrict__ p0, int ld) {
  int lane = threadIdx.x & 31;
  const bf16_t* p = p0 + (lane & 15) * ld + ((lane >> 4) << 3);
  union { v16bf v; uint4 q[2]; } t;
  t.q[0] = *(const uint4*)(p);
  t.q[1] = *(const uint4*)(p + 16);
  return t.v;
}

// ---------- prep kernels ----------

// w: K x N row-major fp32  ->  wt: N x K row-major bf16
__global__ void k_transpose(const float* __restrict__ w, bf16_t* __restrict__ wt,
                            int K, int N) {
  int i = blockIdx.x * blockDim.x + threadIdx.x;
  if (i >= K * N) return;
  int n = i / K, kk = i - n * K;
  wt[i] = (bf16_t)w[kk * N + n];
}

__global__ void k_zero(uint4* __restrict__ p, size_t n4) {
  size_t i = (size_t)blockIdx.x * blockDim.x + threadIdx.x;
  if (i < n4) { uint4 z = {0u, 0u, 0u, 0u}; p[i] = z; }
}

// ---------- LayerNorm kernels (one wave per token) ----------

__global__ void k_ln_shift(const float* __restrict__ x, const float* __restrict__ gw,
                           const float* __restrict__ gb, bf16_t* __restrict__ out) {
  int gt = (blockIdx.x * blockDim.x + threadIdx.x) >> 5;
  int lane = threadIdx.x & 31;
  if (gt >= TOK) return;
  int win = gt / 49, tok = gt - win * 49;
  int img = win >> 6, wr = win & 63;
  int wh = wr >> 3, ww = wr & 7;
  int sh = wh * 7 + tok / 7;
  int sw = ww * 7 + tok % 7;
  int oh = sh + 3; if (oh >= 56) oh -= 56;
  int ow = sw + 3; if (ow >= 56) ow -= 56;
  const float* row = x + ((size_t)img * 3136 + oh * 56 + ow) * DIMC;
  float v[6]; float s = 0.f;
#pragma unroll
  for (int i = 0; i < 6; i++) { v[i] = row[lane + 32 * i]; s += v[i]; }
#pragma unroll
  for (int m = 16; m >= 1; m >>= 1) s += __shfl_xor(s, m, 32);
  float mu = s * (1.f / 192.f);
  float var = 0.f;
#pragma unroll
  for (int i = 0; i < 6; i++) { float d = v[i] - mu; var += d * d; }
#pragma unroll
  for (int m = 16; m >= 1; m >>= 1) var += __shfl_xor(var, m, 32);
  float rs = rsqrtf(var * (1.f / 192.f) + 1e-5f);
  bf16_t* orow = out + (size_t)gt * DIMC;
#pragma unroll
  for (int i = 0; i < 6; i++) {
    int c = lane + 32 * i;
    orow[c] = (bf16_t)((v[i] - mu) * rs * gw[c] + gb[c]);
  }
}

__global__ void k_ln2(const float* __restrict__ xin, const float* __restrict__ gw,
                      const float* __restrict__ gb, bf16_t* __restrict__ out) {
  int gt = (blockIdx.x * blockDim.x + threadIdx.x) >> 5;
  int lane = threadIdx.x & 31;
  if (gt >= TOK) return;
  const float* row = xin + (size_t)gt * DIMC;
  float v[6]; float s = 0.f;
#pragma unroll
  for (int i = 0; i < 6; i++) { v[i] = row[lane + 32 * i]; s += v[i]; }
#pragma unroll
  for (int m = 16; m >= 1; m >>= 1) s += __shfl_xor(s, m, 32);
  float mu = s * (1.f / 192.f);
  float var = 0.f;
#pragma unroll
  for (int i = 0; i < 6; i++) { float d = v[i] - mu; var += d * d; }
#pragma unroll
  for (int m = 16; m >= 1; m >>= 1) var += __shfl_xor(var, m, 32);
  float rs = rsqrtf(var * (1.f / 192.f) + 1e-5f);
  bf16_t* orow = out + (size_t)gt * DIMC;
#pragma unroll
  for (int i = 0; i < 6; i++) {
    int c = lane + 32 * i;
    orow[c] = (bf16_t)((v[i] - mu) * rs * gw[c] + gb[c]);
  }
}

// ---------- register-blocked GEMM kernels ----------
// Each wave computes a 64x32 output block (4x2 tiles of 16x16), K streamed in
// 32-wide WMMA steps: per step 4 A-tile + 2 B-tile loads -> 8 WMMAs.

// QKV: (100352 x 192) @ (192 x 576); scatter into q/k/vT (pad pre-zeroed).
__global__ void k_gemm_qkv(const bf16_t* __restrict__ A, const bf16_t* __restrict__ Wt,
                           const float* __restrict__ bias, bf16_t* __restrict__ q,
                           bf16_t* __restrict__ k, bf16_t* __restrict__ vt) {
  const int NBLK = 18;   // 36 n-tiles / 2
  int wid = (blockIdx.x * blockDim.x + threadIdx.x) >> 5;
  if (wid >= 1568 * NBLK) return;
  int mb = wid / NBLK, nb = wid - mb * NBLK;
  int lane = threadIdx.x & 31;
  v8f acc[4][2] = {};
  for (int s = 0; s < 6; s++) {
    v16bf a[4], b[2];
#pragma unroll
    for (int mi = 0; mi < 4; mi++)
      a[mi] = load_tile16x32(A + (size_t)(mb * 4 + mi) * 16 * DIMC + s * 32, DIMC);
#pragma unroll
    for (int ni = 0; ni < 2; ni++)
      b[ni] = load_tile16x32(Wt + (size_t)(nb * 2 + ni) * 16 * DIMC + s * 32, DIMC);
#pragma unroll
    for (int mi = 0; mi < 4; mi++)
#pragma unroll
      for (int ni = 0; ni < 2; ni++)
        acc[mi][ni] = wmma_bf16(a[mi], b[ni], acc[mi][ni]);
  }
  int nl = lane & 15;
  int mo = (lane >> 4) << 3;
#pragma unroll
  for (int ni = 0; ni < 2; ni++) {
    int nc = (nb * 2 + ni) * 16 + nl;
    float bi = bias[nc];
    int which = nc / DIMC;
    int cc = nc - which * DIMC;
    int head = cc >> 5, d = cc & 31;
#pragma unroll
    for (int mi = 0; mi < 4; mi++) {
      int mbase = (mb * 4 + mi) * 16 + mo;
#pragma unroll
      for (int j = 0; j < 8; j++) {
        int r = mbase + j;
        int win = r / 49, tok = r - win * 49;
        float val = acc[mi][ni][j] + bi;
        size_t hb = (size_t)(win * HEADS + head);
        if (which == 0)      q[(hb * NPAD + tok) * HD + d]  = (bf16_t)(val * SCALE_Q);
        else if (which == 1) k[(hb * NPAD + tok) * HD + d]  = (bf16_t)val;
        else                 vt[(hb * HD + d) * NPAD + tok] = (bf16_t)val;
      }
    }
  }
}

__device__ __forceinline__ int seg3(int u) { return u < 49 ? 0 : (u < 53 ? 1 : 2); }

// Attention: one block (4 waves) per window-head; wave = 16-row strip.
__global__ void k_attn(const bf16_t* __restrict__ q, const bf16_t* __restrict__ k,
                       const bf16_t* __restrict__ vt, const float* __restrict__ rpb,
                       bf16_t* __restrict__ ao) {
  __shared__ float rpbs[169];
  __shared__ __align__(16) bf16_t pbuf[4][16 * NPAD];
  int bh = blockIdx.x;               // = win*HEADS + head
  int win = bh / HEADS, head = bh - win * HEADS;
  int wr = win & 63, wh = wr >> 3, ww = wr & 7;
  for (int i = threadIdx.x; i < 169; i += 128) rpbs[i] = rpb[i * HEADS + head];
  __syncthreads();
  int wave = threadIdx.x >> 5, lane = threadIdx.x & 31;
  const bf16_t* qb = q  + (size_t)bh * (NPAD * HD);
  const bf16_t* kb = k  + (size_t)bh * (NPAD * HD);
  const bf16_t* vb = vt + (size_t)bh * (HD * NPAD);
  int mt = wave;

  // S = Q_strip (16x32) @ K^T  -> 4 tiles of 16x16
  v16bf a = load_tile16x32(qb + mt * 16 * HD, HD);
  float sv[4][8];
  for (int nt = 0; nt < 4; nt++) {
    v16bf b = load_tile16x32(kb + nt * 16 * HD, HD);
    v8f s = {};
    s = wmma_bf16(a, b, s);
#pragma unroll
    for (int j = 0; j < 8; j++) sv[nt][j] = s[j];
  }
  int nl = lane & 15;
  int mo = (lane >> 4) << 3;

  // relative-position bias + shift mask (computed analytically)
  for (int nt = 0; nt < 4; nt++) {
    int n = nt * 16 + nl;              // key token
    bool nvalid = (n < 49);
    int ihk = 0, iwk = 0, clsk = 0;
    if (nvalid) {
      ihk = n / 7; iwk = n - ihk * 7;
      clsk = seg3(wh * 7 + ihk) * 3 + seg3(ww * 7 + iwk);
    }
#pragma unroll
    for (int j = 0; j < 8; j++) {
      int m = mt * 16 + mo + j;        // query token
      if (!nvalid || m >= 49) { sv[nt][j] = -1e30f; continue; }
      int ihq = m / 7, iwq = m - ihq * 7;
      int idx = (ihq - ihk + 6) * 13 + (iwq - iwk + 6);
      int clsq = seg3(wh * 7 + ihq) * 3 + seg3(ww * 7 + iwq);
      sv[nt][j] += rpbs[idx] + ((clsq != clsk) ? -100.f : 0.f);
    }
  }

  // softmax per row (each row lives in 16 lanes x 4 tiles)
#pragma unroll
  for (int j = 0; j < 8; j++) {
    float mx = fmaxf(fmaxf(sv[0][j], sv[1][j]), fmaxf(sv[2][j], sv[3][j]));
#pragma unroll
    for (int d = 8; d >= 1; d >>= 1) mx = fmaxf(mx, __shfl_xor(mx, d, 32));
    float sm = 0.f;
#pragma unroll
    for (int nt = 0; nt < 4; nt++) { float e = __expf(sv[nt][j] - mx); sv[nt][j] = e; sm += e; }
#pragma unroll
    for (int d = 8; d >= 1; d >>= 1) sm += __shfl_xor(sm, d, 32);
    float inv = 1.f / sm;
#pragma unroll
    for (int nt = 0; nt < 4; nt++) sv[nt][j] *= inv;
  }

  // round-trip P through LDS to get it into A-layout
  bf16_t* pw = &pbuf[wave][0];
  for (int nt = 0; nt < 4; nt++) {
    int n = nt * 16 + nl;
#pragma unroll
    for (int j = 0; j < 8; j++) pw[(mo + j) * NPAD + n] = (bf16_t)sv[nt][j];
  }
  __syncthreads();

  // O = P (16x64) @ V (64x32), V stored transposed (32x64)
  for (int nt2 = 0; nt2 < 2; nt2++) {
    v8f o = {};
    for (int ks = 0; ks < 2; ks++) {
      v16bf pa  = load_tile16x32(pw + ks * 32, NPAD);
      v16bf vbm = load_tile16x32(vb + nt2 * 16 * NPAD + ks * 32, NPAD);
      o = wmma_bf16(pa, vbm, o);
    }
    int dcol = nt2 * 16 + nl;
#pragma unroll
    for (int j = 0; j < 8; j++) {
      int tok = mt * 16 + mo + j;
      if (tok < 49)
        ao[((size_t)win * 49 + tok) * DIMC + head * HD + dcol] = (bf16_t)o[j];
    }
  }
}

// Proj GEMM fused with window-reverse + reverse shift + shortcut add -> fp32 out.
__global__ void k_gemm_proj(const bf16_t* __restrict__ A, const bf16_t* __restrict__ Wt,
                            const float* __restrict__ bias, const float* __restrict__ xin,
                            float* __restrict__ out) {
  const int NBLK = 6;    // 12 n-tiles / 2
  int wid = (blockIdx.x * blockDim.x + threadIdx.x) >> 5;
  if (wid >= 1568 * NBLK) return;
  int mb = wid / NBLK, nb = wid - mb * NBLK;
  int lane = threadIdx.x & 31;
  v8f acc[4][2] = {};
  for (int s = 0; s < 6; s++) {
    v16bf a[4], b[2];
#pragma unroll
    for (int mi = 0; mi < 4; mi++)
      a[mi] = load_tile16x32(A + (size_t)(mb * 4 + mi) * 16 * DIMC + s * 32, DIMC);
#pragma unroll
    for (int ni = 0; ni < 2; ni++)
      b[ni] = load_tile16x32(Wt + (size_t)(nb * 2 + ni) * 16 * DIMC + s * 32, DIMC);
#pragma unroll
    for (int mi = 0; mi < 4; mi++)
#pragma unroll
      for (int ni = 0; ni < 2; ni++)
        acc[mi][ni] = wmma_bf16(a[mi], b[ni], acc[mi][ni]);
  }
  int nl = lane & 15;
  int mo = (lane >> 4) << 3;
#pragma unroll
  for (int ni = 0; ni < 2; ni++) {
    int nc = (nb * 2 + ni) * 16 + nl;
    float bi = bias[nc];
#pragma unroll
    for (int mi = 0; mi < 4; mi++) {
      int mbase = (mb * 4 + mi) * 16 + mo;
#pragma unroll
      for (int j = 0; j < 8; j++) {
        int r = mbase + j;
        int win = r / 49, tok = r - win * 49;
        int img = win >> 6, wr = win & 63;
        int wh = wr >> 3, ww = wr & 7;
        int sh = wh * 7 + tok / 7;
        int sw = ww * 7 + tok % 7;
        int oh = sh + 3; if (oh >= 56) oh -= 56;
        int ow = sw + 3; if (ow >= 56) ow -= 56;
        size_t oi = ((size_t)img * 3136 + oh * 56 + ow) * DIMC + nc;
        out[oi] = xin[oi] + acc[mi][ni][j] + bi;
      }
    }
  }
}

// FC1 GEMM + exact GELU -> bf16 (100352 x 768)
__global__ void k_gemm_fc1(const bf16_t* __restrict__ A, const bf16_t* __restrict__ Wt,
                           const float* __restrict__ bias, bf16_t* __restrict__ out) {
  const int NBLK = 24;   // 48 n-tiles / 2
  int wid = (blockIdx.x * blockDim.x + threadIdx.x) >> 5;
  if (wid >= 1568 * NBLK) return;
  int mb = wid / NBLK, nb = wid - mb * NBLK;
  int lane = threadIdx.x & 31;
  v8f acc[4][2] = {};
  for (int s = 0; s < 6; s++) {
    v16bf a[4], b[2];
#pragma unroll
    for (int mi = 0; mi < 4; mi++)
      a[mi] = load_tile16x32(A + (size_t)(mb * 4 + mi) * 16 * DIMC + s * 32, DIMC);
#pragma unroll
    for (int ni = 0; ni < 2; ni++)
      b[ni] = load_tile16x32(Wt + (size_t)(nb * 2 + ni) * 16 * DIMC + s * 32, DIMC);
#pragma unroll
    for (int mi = 0; mi < 4; mi++)
#pragma unroll
      for (int ni = 0; ni < 2; ni++)
        acc[mi][ni] = wmma_bf16(a[mi], b[ni], acc[mi][ni]);
  }
  int nl = lane & 15;
  int mo = (lane >> 4) << 3;
#pragma unroll
  for (int ni = 0; ni < 2; ni++) {
    int nc = (nb * 2 + ni) * 16 + nl;
    float bi = bias[nc];
#pragma unroll
    for (int mi = 0; mi < 4; mi++) {
      int mbase = (mb * 4 + mi) * 16 + mo;
#pragma unroll
      for (int j = 0; j < 8; j++) {
        int r = mbase + j;
        float v = acc[mi][ni][j] + bi;
        float g = 0.5f * v * (1.f + erff(v * 0.70710678118654752f));
        out[(size_t)r * 768 + nc] = (bf16_t)g;
      }
    }
  }
}

// FC2 GEMM, accumulate into fp32 residual buffer (final output)
__global__ void k_gemm_fc2(const bf16_t* __restrict__ A, const bf16_t* __restrict__ Wt,
                           const float* __restrict__ bias, float* __restrict__ out) {
  const int NBLK = 6;    // 12 n-tiles / 2
  int wid = (blockIdx.x * blockDim.x + threadIdx.x) >> 5;
  if (wid >= 1568 * NBLK) return;
  int mb = wid / NBLK, nb = wid - mb * NBLK;
  int lane = threadIdx.x & 31;
  v8f acc[4][2] = {};
  for (int s = 0; s < 24; s++) {
    v16bf a[4], b[2];
#pragma unroll
    for (int mi = 0; mi < 4; mi++)
      a[mi] = load_tile16x32(A + (size_t)(mb * 4 + mi) * 16 * 768 + s * 32, 768);
#pragma unroll
    for (int ni = 0; ni < 2; ni++)
      b[ni] = load_tile16x32(Wt + (size_t)(nb * 2 + ni) * 16 * 768 + s * 32, 768);
#pragma unroll
    for (int mi = 0; mi < 4; mi++)
#pragma unroll
      for (int ni = 0; ni < 2; ni++)
        acc[mi][ni] = wmma_bf16(a[mi], b[ni], acc[mi][ni]);
  }
  int nl = lane & 15;
  int mo = (lane >> 4) << 3;
#pragma unroll
  for (int ni = 0; ni < 2; ni++) {
    int nc = (nb * 2 + ni) * 16 + nl;
    float bi = bias[nc];
#pragma unroll
    for (int mi = 0; mi < 4; mi++) {
      int mbase = (mb * 4 + mi) * 16 + mo;
#pragma unroll
      for (int j = 0; j < 8; j++) {
        int r = mbase + j;
        size_t oi = (size_t)r * DIMC + nc;
        out[oi] = out[oi] + acc[mi][ni][j] + bi;
      }
    }
  }
}

// ---------- launch ----------

extern "C" void kernel_launch(void* const* d_in, const int* in_sizes, int n_in,
                              void* d_out, int out_size, void* d_ws, size_t ws_size,
                              hipStream_t stream) {
  const float* x      = (const float*)d_in[0];
  const float* n1w    = (const float*)d_in[1];
  const float* n1b    = (const float*)d_in[2];
  const float* qkv_w  = (const float*)d_in[3];
  const float* qkv_b  = (const float*)d_in[4];
  const float* rpb    = (const float*)d_in[5];
  const float* proj_w = (const float*)d_in[6];
  const float* proj_b = (const float*)d_in[7];
  const float* n2w    = (const float*)d_in[8];
  const float* n2b    = (const float*)d_in[9];
  const float* fc1_w  = (const float*)d_in[10];
  const float* fc1_b  = (const float*)d_in[11];
  const float* fc2_w  = (const float*)d_in[12];
  const float* fc2_b  = (const float*)d_in[13];
  float* out = (float*)d_out;

  // workspace layout (bf16), with phase-based aliasing
  bf16_t* Wqkv  = (bf16_t*)d_ws;                       // 576*192
  bf16_t* Wproj = Wqkv  + 576 * 192;                   // 192*192
  bf16_t* Wfc1  = Wproj + 192 * 192;                   // 768*192
  bf16_t* Wfc2  = Wfc1  + 768 * 192;                   // 192*768
  bf16_t* hw    = Wfc2  + 192 * 768;                   // 100352*192 (LN1 out; reused for LN2 out)
  bf16_t* ao    = hw + (size_t)TOK * DIMC;             // 100352*192 (attn out)
  bf16_t* qb    = ao + (size_t)TOK * DIMC;             // 2048*6*64*32
  bf16_t* kb    = qb + (size_t)NWIN * HEADS * NPAD * HD;
  bf16_t* vtb   = kb + (size_t)NWIN * HEADS * NPAD * HD;
  bf16_t* g1    = qb;                                  // alias: 100352*768 after attention

  // weight transposes (fp32 -> bf16, N x K)
  k_transpose<<<(576 * 192 + 255) / 256, 256, 0, stream>>>(qkv_w, Wqkv, 192, 576);
  k_transpose<<<(192 * 192 + 255) / 256, 256, 0, stream>>>(proj_w, Wproj, 192, 192);
  k_transpose<<<(768 * 192 + 255) / 256, 256, 0, stream>>>(fc1_w, Wfc1, 192, 768);
  k_transpose<<<(192 * 768 + 255) / 256, 256, 0, stream>>>(fc2_w, Wfc2, 768, 192);

  // zero q/k/vT (covers the 49->64 token padding)
  size_t zero4 = (size_t)3 * NWIN * HEADS * NPAD * HD / 8;   // uint4 count
  k_zero<<<(unsigned)((zero4 + 255) / 256), 256, 0, stream>>>((uint4*)qb, zero4);

  // LN1 + shift + partition
  k_ln_shift<<<TOK / 8, 256, 0, stream>>>(x, n1w, n1b, hw);
  // QKV projection: 1568 m-blocks * 18 n-blocks waves
  k_gemm_qkv<<<1568 * 18 / 8, 256, 0, stream>>>(hw, Wqkv, qkv_b, qb, kb, vtb);
  // windowed attention
  k_attn<<<NWIN * HEADS, 128, 0, stream>>>(qb, kb, vtb, rpb, ao);
  // proj + reverse + shortcut -> d_out (fp32 residual)
  k_gemm_proj<<<1568 * 6 / 8, 256, 0, stream>>>(ao, Wproj, proj_b, x, out);
  // MLP
  k_ln2<<<TOK / 8, 256, 0, stream>>>(out, n2w, n2b, hw);
  k_gemm_fc1<<<1568 * 24 / 8, 256, 0, stream>>>(hw, Wfc1, fc1_b, g1);
  k_gemm_fc2<<<1568 * 6 / 8, 256, 0, stream>>>(g1, Wfc2, fc2_b, out);
}
